// Performers_41214506172541
// MI455X (gfx1250) — matrix operand order repaired and checked by reading, not compiled, not run on hardware
//
#include <hip/hip_runtime.h>
#include <hip/hip_bf16.h>

// Problem constants (from reference): B=2, L=1024, E=512, H=8, DH=64, M=128
#define PB 2
#define PL 1024
#define PE 512
#define PH 8
#define PDH 64
#define PM 128
#define ROWS (PB * PL)          // 2048
#define SCALE_DATA 0.3535533905932738f   // 64^-0.25
#define RATIO 0.08838834764831845f       // 128^-0.5
#define FEPS 1e-5f

typedef __attribute__((ext_vector_type(16))) _Float16 v16h;
typedef __attribute__((ext_vector_type(8)))  float    v8f;
typedef unsigned int v4u __attribute__((ext_vector_type(4)));
typedef int          v8i __attribute__((ext_vector_type(8)));
typedef int          v4i __attribute__((ext_vector_type(4)));

#if defined(__has_builtin)
#if __has_builtin(__builtin_amdgcn_tensor_load_to_lds) && __has_builtin(__builtin_amdgcn_s_wait_tensorcnt)
#define HAVE_TDM 1
#endif
#endif

// ---------------------------------------------------------------------------
// WMMA helpers (CDNA5 gfx1250, wave32). D = A(16x32 f16) * B(32x16 f16) + C(f32)
// ---------------------------------------------------------------------------
__device__ __forceinline__ v8f wmma_f16(v16h a, v16h b, v8f c) {
  return __builtin_amdgcn_wmma_f32_16x16x32_f16(
      /*neg_a=*/false, a, /*neg_b=*/false, b,
      /*c_mod=*/(short)0, c, /*reuse_a=*/false, /*reuse_b=*/false);
}

// A fragment (16x32 f16), ISA 7.12.2 layout. element(M,K)=base[M*sM + K*sK]
__device__ __forceinline__ v16h load_frag_a(const _Float16* base, int strideM, int strideK) {
  int lane  = threadIdx.x & 31;
  int mrow  = lane & 15;
  int khalf = (lane >> 4) << 3;   // 0 or 8
  v16h r;
#pragma unroll
  for (int v = 0; v < 8; ++v) {
    int kb = ((v & 4) << 2) + khalf + ((v & 3) << 1);
    r[2 * v]     = base[mrow * strideM + (kb + 0) * strideK];
    r[2 * v + 1] = base[mrow * strideM + (kb + 1) * strideK];
  }
  return r;
}

// B fragment (32x16 f16). element(K,N)=base[K*sK + N*sN]
__device__ __forceinline__ v16h load_frag_b(const _Float16* base, int strideK, int strideN) {
  int lane = threadIdx.x & 31;
  int n    = lane & 15;
  int kh   = (lane >> 4) << 3;    // 0 or 8
  v16h r;
#pragma unroll
  for (int v = 0; v < 8; ++v) {
    int kb = ((v & 4) << 2) + kh + ((v & 3) << 1);
    r[2 * v]     = base[(kb + 0) * strideK + n * strideN];
    r[2 * v + 1] = base[(kb + 1) * strideK + n * strideN];
  }
  return r;
}

// C/D f32 16x16: VGPR r -> M = r + (lane>=16 ? 8 : 0), N = lane&15.

#ifdef HAVE_TDM
// ---------------------------------------------------------------------------
// Tensor Data Mover: 2D tile load (f16 elements) global -> LDS, with LDS row
// padding to match our padded LDS pitch.  D# per ISA 8.3/8.4.
// 6-arg builtin form (clang-23 / therock-10.0 headers): groups + zero operands
// + cache policy.
// ---------------------------------------------------------------------------
__device__ __forceinline__ void tdm_load_2d_f16(
    unsigned lds_off, const void* gaddr,
    unsigned tensor_d0, unsigned tensor_d1, unsigned stride0,
    unsigned tile0, unsigned tile1,
    unsigned pad_interval_code, unsigned pad_amount_code) {
  unsigned long long ga = (unsigned long long)(uintptr_t)gaddr;
  v4u g0;
  g0[0] = 1u;                                        // count=1 (valid user D#)
  g0[1] = lds_off;                                   // lds_addr (bytes)
  g0[2] = (unsigned)ga;                              // global_addr[31:0]
  g0[3] = (unsigned)((ga >> 32) & 0x01FFFFFFu) | 0x80000000u; // addr[56:32] | type=2
  v8i g1;
  g1[0] = (int)((1u << 16)                           // data_size = 2 bytes
              | (1u << 20)                           // pad_enable
              | (pad_interval_code << 22)
              | (pad_amount_code << 25));
  g1[1] = (int)((tensor_d0 & 0xFFFFu) << 16);        // [47:32]=atomic addr 0, dim0.lo16
  g1[2] = (int)((tensor_d0 >> 16) | ((tensor_d1 & 0xFFFFu) << 16));
  g1[3] = (int)((tensor_d1 >> 16) | (tile0 << 16));  // dim1.hi16 | tile_dim0
  g1[4] = (int)(tile1 & 0xFFFFu);                    // tile_dim1 ; tile_dim2=0
  g1[5] = (int)stride0;                              // tensor_dim0_stride[31:0]
  g1[6] = 0;                                         // stride hi, dim1_stride lo
  g1[7] = 0;
  v4i gz4 = {0, 0, 0, 0};
  v8i gz8 = {0, 0, 0, 0, 0, 0, 0, 0};
  __builtin_amdgcn_tensor_load_to_lds(g0, g1, gz4, gz4, gz8, 0);
}
#endif

// ---------------------------------------------------------------------------
// K1/K5: Y(Rows x N) = X(Rows x K) @ W^T + bias.   W is (N x K) row-major.
// 128x128 output tile per 256-thread block (8 waves, each 64x32 of output).
// ---------------------------------------------------------------------------
__global__ __launch_bounds__(256) void gemm_bias_wmma(
    const float* __restrict__ X, const float* __restrict__ W,
    const float* __restrict__ bias, float* __restrict__ Y,
    int Rows, int Kdim, int Ncols) {
  __shared__ _Float16 aT[128][40];   // [row][k]
  __shared__ _Float16 bT[32][136];   // [k][n]
  int t    = threadIdx.x;
  int wave = t >> 5;
  int row0 = blockIdx.x * 128;
  int col0 = blockIdx.y * 128;
  int wr = wave >> 2;
  int wc = wave & 3;

  v8f acc[4][2];
#pragma unroll
  for (int mi = 0; mi < 4; ++mi)
#pragma unroll
    for (int j = 0; j < 2; ++j)
#pragma unroll
      for (int r = 0; r < 8; ++r) acc[mi][j][r] = 0.f;

  for (int k0 = 0; k0 < Kdim; k0 += 32) {
    __syncthreads();
    {   // A tile: 128x32
      int r = t >> 1, c0 = (t & 1) * 16;
      const float* src = X + (size_t)(row0 + r) * Kdim + k0 + c0;
#pragma unroll
      for (int i = 0; i < 16; ++i) aT[r][c0 + i] = (_Float16)src[i];
      if (k0 + 32 < Kdim) __builtin_prefetch(src + 32, 0, 0);  // global_prefetch_b8
    }
    {   // B tile: bT[k][n] = W[(col0+n)*Kdim + k0+k]
      int n = t >> 1, kk0 = (t & 1) * 16;
      const float* src = W + (size_t)(col0 + n) * Kdim + k0 + kk0;
#pragma unroll
      for (int i = 0; i < 16; ++i) bT[kk0 + i][n] = (_Float16)src[i];
      if (k0 + 32 < Kdim) __builtin_prefetch(src + 32, 0, 0);
    }
    __syncthreads();
    v16h bf0 = load_frag_b(&bT[0][wc * 32], 136, 1);
    v16h bf1 = load_frag_b(&bT[0][wc * 32 + 16], 136, 1);
#pragma unroll
    for (int mi = 0; mi < 4; ++mi) {
      v16h af = load_frag_a(&aT[wr * 64 + mi * 16][0], 40, 1);
      acc[mi][0] = wmma_f16(af, bf0, acc[mi][0]);
      acc[mi][1] = wmma_f16(af, bf1, acc[mi][1]);
    }
  }
  int lane = t & 31, mh = (lane >> 4) << 3, nl = lane & 15;
#pragma unroll
  for (int mi = 0; mi < 4; ++mi)
#pragma unroll
    for (int j = 0; j < 2; ++j)
#pragma unroll
      for (int r = 0; r < 8; ++r) {
        int gr = row0 + wr * 64 + mi * 16 + r + mh;
        int gc = col0 + wc * 32 + j * 16 + nl;
        Y[(size_t)gr * Ncols + gc] = acc[mi][j][r] + bias[gc];
      }
}

// ---------------------------------------------------------------------------
// K2: dash[bl,h,m] = (scale*data[bl,h*64+:64]) @ proj^T, diag = 0.5*|scaled|^2
// ---------------------------------------------------------------------------
__global__ __launch_bounds__(128) void dash_diag_wmma(
    const float* __restrict__ data,   // (ROWS, 512)
    const float* __restrict__ proj,   // (128, 64)
    float* __restrict__ dash,         // (ROWS, H, M)
    float* __restrict__ diag) {       // (ROWS, H)
  __shared__ _Float16 aT[32][72];
  __shared__ _Float16 pT[64][136];
  int t = threadIdx.x;
  int h = blockIdx.y;
  int row0 = blockIdx.x * 32;

  for (int d = 0; d < 64; ++d) pT[d][t] = (_Float16)proj[t * 64 + d];
  {
    int r = t >> 2, c0 = (t & 3) * 16;
    const float* src = data + (size_t)(row0 + r) * PE + h * PDH + c0;
#pragma unroll
    for (int i = 0; i < 16; ++i) aT[r][c0 + i] = (_Float16)(src[i] * SCALE_DATA);
  }
  __syncthreads();
  if (t < 32) {
    const float* src = data + (size_t)(row0 + t) * PE + h * PDH;
    float s = 0.f;
    for (int d = 0; d < PDH; ++d) { float x = src[d] * SCALE_DATA; s += x * x; }
    diag[(size_t)(row0 + t) * PH + h] = 0.5f * s;
  }
  int wave = t >> 5;
  v8f acc[2][2];
#pragma unroll
  for (int mi = 0; mi < 2; ++mi)
#pragma unroll
    for (int j = 0; j < 2; ++j)
#pragma unroll
      for (int r = 0; r < 8; ++r) acc[mi][j][r] = 0.f;

  for (int ks = 0; ks < 2; ++ks) {
    v16h b0 = load_frag_b(&pT[ks * 32][wave * 32], 136, 1);
    v16h b1 = load_frag_b(&pT[ks * 32][wave * 32 + 16], 136, 1);
#pragma unroll
    for (int mi = 0; mi < 2; ++mi) {
      v16h a = load_frag_a(&aT[mi * 16][ks * 32], 72, 1);
      acc[mi][0] = wmma_f16(a, b0, acc[mi][0]);
      acc[mi][1] = wmma_f16(a, b1, acc[mi][1]);
    }
  }
  int lane = t & 31, mh = (lane >> 4) << 3, nl = lane & 15;
#pragma unroll
  for (int mi = 0; mi < 2; ++mi)
#pragma unroll
    for (int j = 0; j < 2; ++j)
#pragma unroll
      for (int r = 0; r < 8; ++r) {
        int gr = row0 + mi * 16 + r + mh;
        int gm = wave * 32 + j * 16 + nl;
        dash[((size_t)gr * PH + h) * PM + gm] = acc[mi][j][r];
      }
}

// ---------------------------------------------------------------------------
// K3: max reductions + feature transform (writes f16 features for the scan)
// ---------------------------------------------------------------------------
__global__ void rowmax_kernel(const float* __restrict__ dash, float* __restrict__ mx, int nrows) {
  int r = blockIdx.x * blockDim.x + threadIdx.x;
  if (r < nrows) {
    float m = -3.402823466e38f;
    const float* p = dash + (size_t)r * PM;
    for (int i = 0; i < PM; ++i) m = fmaxf(m, p[i]);
    mx[r] = m;
  }
}

__global__ __launch_bounds__(256) void maxk_kernel(const float* __restrict__ dashK,
                                                   float* __restrict__ mxK) {
  int b = blockIdx.x >> 3, h = blockIdx.x & 7;
  float m = -3.402823466e38f;
  for (int idx = threadIdx.x; idx < PL * PM; idx += 256) {
    int l = idx >> 7, mm = idx & 127;
    m = fmaxf(m, dashK[(((size_t)b * PL + l) * PH + h) * PM + mm]);
  }
  __shared__ float red[256];
  red[threadIdx.x] = m;
  __syncthreads();
  for (int s = 128; s > 0; s >>= 1) {
    if (threadIdx.x < s) red[threadIdx.x] = fmaxf(red[threadIdx.x], red[threadIdx.x + s]);
    __syncthreads();
  }
  if (threadIdx.x == 0) mxK[blockIdx.x] = red[0];
}

__global__ void apply_features_f16(const float* __restrict__ dash,
                                   const float* __restrict__ diag,
                                   const float* __restrict__ mx, int per_row_mx, int n,
                                   _Float16* __restrict__ feat) {
  int i = blockIdx.x * blockDim.x + threadIdx.x;
  if (i < n) {
    int row = i >> 7;    // bl*H + h
    float m = per_row_mx ? mx[row] : mx[(row >> 13) * PH + (row & 7)];
    feat[i] = (_Float16)(RATIO * (expf(dash[i] - diag[row] - m) + FEPS));
  }
}

__global__ void f32_to_f16(const float* __restrict__ src, _Float16* __restrict__ dst, int n) {
  int i = blockIdx.x * blockDim.x + threadIdx.x;
  if (i < n) dst[i] = (_Float16)src[i];
}

// ---------------------------------------------------------------------------
// K4: chunked causal FAVOR scan, one workgroup per (b,h), 4 waves; each wave
// owns a 16-wide DH slice of the 128x64 KV state (8 f32 WMMA accumulators).
// Chunk T=32:  P = Qc Kc^T (masked),  O = P Vc + Qc KV,  KV += Kc^T Vc,
//              den = Qc*ksum + rowsum(P_masked).
// Tiles staged into LDS via the Tensor Data Mover when available.
// ---------------------------------------------------------------------------
__global__ __launch_bounds__(128) void favor_scan(
    const _Float16* __restrict__ qp,    // (ROWS, H, M) f16
    const _Float16* __restrict__ kp,    // (ROWS, H, M) f16
    const _Float16* __restrict__ vp,    // (ROWS, 512)  f16
    float* __restrict__ outAttn) {      // (ROWS, 512)  f32
  __shared__ _Float16 qS[32][136];   // row pitch = 68 DWORDs = 64 data + 4 pad
  __shared__ _Float16 kS[32][136];
  __shared__ _Float16 vS[32][72];    // row pitch = 36 DWORDs = 32 data + 4 pad
  __shared__ _Float16 kvS[128][72];
  __shared__ _Float16 pS[32][40];
  __shared__ float ksum[128];
  __shared__ float den[32];

  int t    = threadIdx.x;
  int wave = t >> 5;
  int lane = t & 31, mh = (lane >> 4) << 3, nl = lane & 15;
  int b = blockIdx.x >> 3, h = blockIdx.x & 7;
  size_t rowBase = (size_t)b * PL;

  v8f kvacc[8];
#pragma unroll
  for (int kt = 0; kt < 8; ++kt)
#pragma unroll
    for (int r = 0; r < 8; ++r) kvacc[kt][r] = 0.f;
  ksum[t] = 0.f;

  for (int c = 0; c < PL / 32; ++c) {
    int l0 = c * 32;
    __syncthreads();
#ifdef HAVE_TDM
    if (wave == 0) {
      // TDM: q/k tiles 32x128 f16 (pad 4 DWORDs per 64), v tile 32x64 f16
      const _Float16* qg = qp + ((rowBase + l0) * PH + h) * PM;
      const _Float16* kg = kp + ((rowBase + l0) * PH + h) * PM;
      const _Float16* vg = vp + (rowBase + l0) * PE + h * PDH;
      tdm_load_2d_f16((unsigned)(uintptr_t)&qS[0][0], qg, PM, 32, PH * PM, PM, 32, 5, 3);
      tdm_load_2d_f16((unsigned)(uintptr_t)&kS[0][0], kg, PM, 32, PH * PM, PM, 32, 5, 3);
      tdm_load_2d_f16((unsigned)(uintptr_t)&vS[0][0], vg, PDH, 32, PE, PDH, 32, 4, 3);
      __builtin_amdgcn_s_wait_tensorcnt(0);
    }
#else
    {   // fallback: manual copy (no conversion, data already f16)
      int r = t >> 2, m0 = (t & 3) * 32;
      const _Float16* qs = qp + ((rowBase + l0 + r) * PH + h) * PM + m0;
      const _Float16* ks = kp + ((rowBase + l0 + r) * PH + h) * PM + m0;
#pragma unroll
      for (int i = 0; i < 32; ++i) { qS[r][m0 + i] = qs[i]; kS[r][m0 + i] = ks[i]; }
      int d0 = (t & 3) * 16;
      const _Float16* vs = vp + (rowBase + l0 + r) * PE + h * PDH + d0;
#pragma unroll
      for (int i = 0; i < 16; ++i) vS[r][d0 + i] = vs[i];
    }
#endif
    // dump KV state (f32 accum) -> LDS f16 for B-operand use
#pragma unroll
    for (int kt = 0; kt < 8; ++kt)
#pragma unroll
      for (int r = 0; r < 8; ++r)
        kvS[kt * 16 + r + mh][wave * 16 + nl] = (_Float16)kvacc[kt][r];
    __syncthreads();

    // O_inter = Qc (32x128) @ KV slice (128x16)
    v8f oacc[2];
#pragma unroll
    for (int mi = 0; mi < 2; ++mi)
#pragma unroll
      for (int r = 0; r < 8; ++r) oacc[mi][r] = 0.f;
#pragma unroll
    for (int ks = 0; ks < 4; ++ks) {
      v16h bfr = load_frag_b(&kvS[ks * 32][wave * 16], 72, 1);
#pragma unroll
      for (int mi = 0; mi < 2; ++mi) {
        v16h a = load_frag_a(&qS[mi * 16][ks * 32], 136, 1);
        oacc[mi] = wmma_f16(a, bfr, oacc[mi]);
      }
    }
    if (wave == 0) {   // P = Qc @ Kc^T (32x32), masked causal -> pS (f16)
      v8f pacc[2][2];
#pragma unroll
      for (int mi = 0; mi < 2; ++mi)
#pragma unroll
        for (int nj = 0; nj < 2; ++nj)
#pragma unroll
          for (int r = 0; r < 8; ++r) pacc[mi][nj][r] = 0.f;
#pragma unroll
      for (int ks = 0; ks < 4; ++ks) {
        v16h a0 = load_frag_a(&qS[0][ks * 32], 136, 1);
        v16h a1 = load_frag_a(&qS[16][ks * 32], 136, 1);
        v16h b0 = load_frag_b(&kS[0][ks * 32], 1, 136);   // B[m][j] = kS[j][m]
        v16h b1 = load_frag_b(&kS[16][ks * 32], 1, 136);
        pacc[0][0] = wmma_f16(a0, b0, pacc[0][0]);
        pacc[0][1] = wmma_f16(a0, b1, pacc[0][1]);
        pacc[1][0] = wmma_f16(a1, b0, pacc[1][0]);
        pacc[1][1] = wmma_f16(a1, b1, pacc[1][1]);
      }
#pragma unroll
      for (int mi = 0; mi < 2; ++mi)
#pragma unroll
        for (int nj = 0; nj < 2; ++nj)
#pragma unroll
          for (int r = 0; r < 8; ++r) {
            int lrow = mi * 16 + r + mh, jcol = nj * 16 + nl;
            float v = (jcol <= lrow) ? pacc[mi][nj][r] : 0.f;
            pS[lrow][jcol] = (_Float16)v;
          }
    }
    __syncthreads();
    if (t < 32) {   // den_l = q_l . ksum_prefix + rowsum(P_masked)
      float s = 0.f;
      for (int m = 0; m < PM; ++m) s += (float)qS[t][m] * ksum[m];
      for (int j = 0; j <= t; ++j) s += (float)pS[t][j];
      den[t] = s;
    }
    __syncthreads();
    {   // ksum += colsum(Kc)
      float s = ksum[t];
      for (int tt = 0; tt < 32; ++tt) s += (float)kS[tt][t];
      ksum[t] = s;
    }
    {   // O_intra and KV update
      v16h bv = load_frag_b(&vS[0][wave * 16], 72, 1);
#pragma unroll
      for (int mi = 0; mi < 2; ++mi) {
        v16h a = load_frag_a(&pS[mi * 16][0], 40, 1);
        oacc[mi] = wmma_f16(a, bv, oacc[mi]);
      }
#pragma unroll
      for (int kt = 0; kt < 8; ++kt) {
        v16h a = load_frag_a(&kS[0][kt * 16], 1, 136);  // A[m][K] = kS[K][kt*16+m]
        kvacc[kt] = wmma_f16(a, bv, kvacc[kt]);
      }
    }
#pragma unroll
    for (int mi = 0; mi < 2; ++mi)
#pragma unroll
      for (int r = 0; r < 8; ++r) {
        int lrow = mi * 16 + r + mh;
        int d = wave * 16 + nl;
        outAttn[(rowBase + l0 + lrow) * PE + h * PDH + d] = oacc[mi][r] / den[lrow];
      }
  }
}

// ---------------------------------------------------------------------------
// Launch: qkv GEMMs -> dash/diag -> maxes -> f16 features -> TDM+WMMA scan -> W0
// Workspace use: ~45 MB of scratch.
// ---------------------------------------------------------------------------
extern "C" void kernel_launch(void* const* d_in, const int* in_sizes, int n_in,
                              void* d_out, int out_size, void* d_ws, size_t ws_size,
                              hipStream_t stream) {
  const float* query = (const float*)d_in[0];
  const float* key   = (const float*)d_in[1];
  const float* value = (const float*)d_in[2];
  const float* Wq = (const float*)d_in[3];
  const float* Wk = (const float*)d_in[4];
  const float* Wv = (const float*)d_in[5];
  const float* W0 = (const float*)d_in[6];
  const float* bq = (const float*)d_in[7];
  const float* bk = (const float*)d_in[8];
  const float* bv = (const float*)d_in[9];
  const float* b0 = (const float*)d_in[10];
  const float* proj = (const float*)d_in[11];

  float* ws = (float*)d_ws;
  float* qProj = ws;                          // 2048*512
  float* kProj = qProj + ROWS * PE;
  float* vProj = kProj + ROWS * PE;
  float* dashQ = vProj + ROWS * PE;           // 2048*8*128
  float* dashK = dashQ + ROWS * PH * PM;
  float* diagQ = dashK + ROWS * PH * PM;      // 16384
  float* diagK = diagQ + ROWS * PH;
  float* mxQ   = diagK + ROWS * PH;
  float* mxK   = mxQ + ROWS * PH;             // 16 (padded to 64)
  float* attn  = mxK + 64;                    // 2048*512
  _Float16* qp16 = (_Float16*)(attn + ROWS * PE);        // 2048*8*128 halfs
  _Float16* kp16 = qp16 + ROWS * PH * PM;
  _Float16* vp16 = kp16 + ROWS * PH * PM;                // 2048*512 halfs

  dim3 gG(ROWS / 128, PE / 128);
  gemm_bias_wmma<<<gG, 256, 0, stream>>>(query, Wq, bq, qProj, ROWS, PE, PE);
  gemm_bias_wmma<<<gG, 256, 0, stream>>>(key,   Wk, bk, kProj, ROWS, PE, PE);
  gemm_bias_wmma<<<gG, 256, 0, stream>>>(value, Wv, bv, vProj, ROWS, PE, PE);

  dim3 gD(ROWS / 32, PH);
  dash_diag_wmma<<<gD, 128, 0, stream>>>(qProj, proj, dashQ, diagQ);
  dash_diag_wmma<<<gD, 128, 0, stream>>>(kProj, proj, dashK, diagK);

  rowmax_kernel<<<(ROWS * PH + 255) / 256, 256, 0, stream>>>(dashQ, mxQ, ROWS * PH);
  maxk_kernel<<<PB * PH, 256, 0, stream>>>(dashK, mxK);

  int nfeat = ROWS * PH * PM;
  apply_features_f16<<<(nfeat + 255) / 256, 256, 0, stream>>>(dashQ, diagQ, mxQ, 1, nfeat, qp16);
  apply_features_f16<<<(nfeat + 255) / 256, 256, 0, stream>>>(dashK, diagK, mxK, 0, nfeat, kp16);
  f32_to_f16<<<(ROWS * PE + 255) / 256, 256, 0, stream>>>(vProj, vp16, ROWS * PE);

  favor_scan<<<PB * PH, 128, 0, stream>>>(qp16, kp16, vp16, attn);

  gemm_bias_wmma<<<gG, 256, 0, stream>>>(attn, W0, b0, (float*)d_out, ROWS, PE, PE);
}
